// ScaledDotProductAttention_30331059044530
// MI455X (gfx1250) — matrix operand order repaired and checked by reading, not compiled, not run on hardware
//
#include <hip/hip_runtime.h>
#include <hip/hip_bf16.h>

// MI455X / gfx1250, wave32. bf16 WMMA for all GEMMs (f32 accum).
// Flash-style online softmax; attn matrix is a required output (268MB) so raw
// scores are streamed out and normalized in a second pass.
// K/V tiles are DMA'd into double-buffered LDS by the Tensor Data Mover
// (tensor_load_to_lds, TENSORcnt) and shared by all 8 waves of the block.

typedef __attribute__((ext_vector_type(16))) __bf16 v16bf;
typedef __attribute__((ext_vector_type(8)))  float  v8f;
typedef __attribute__((ext_vector_type(4)))  unsigned int u32x4;
typedef __attribute__((ext_vector_type(8)))  int i32x8;
typedef __attribute__((ext_vector_type(4)))  int i32x4;

union AF { v16bf v; unsigned int d[8]; unsigned short u[16]; };
union CF { v8f v; float f[8]; };

__device__ inline unsigned short f2bf(float f) {
  unsigned int x = __builtin_bit_cast(unsigned int, f);
  return (unsigned short)((x + 0x7FFFu + ((x >> 16) & 1u)) >> 16);
}

__device__ inline float rowmax16(float v) {
  #pragma unroll
  for (int off = 1; off < 16; off <<= 1) v = fmaxf(v, __shfl_xor(v, off, 16));
  return v;
}
__device__ inline float rowsum16(float v) {
  #pragma unroll
  for (int off = 1; off < 16; off <<= 1) v += __shfl_xor(v, off, 16);
  return v;
}

// ---------------------------------------------------------------------------
// Tensor Data Mover: build D# (ISA 08_async_tensor §8.3/8.4) and issue a
// 2D bf16 tile load  global -> LDS.  tile_d0/tile_d1 in elements,
// stride0 in elements.  LDS receives tile rows packed contiguously.
// ---------------------------------------------------------------------------
__device__ inline void tdm_load_tile_bf16(unsigned int lds_off, const void* gsrc,
                                          unsigned int tile_d0, unsigned int tile_d1,
                                          unsigned int tensor_d0, unsigned int tensor_d1,
                                          unsigned long long stride0) {
  unsigned long long ga = (unsigned long long)(uintptr_t)gsrc;
  u32x4 g0;
  g0[0] = 1u;                                       // count=1, user mode
  g0[1] = lds_off;                                  // LDS byte address
  g0[2] = (unsigned int)(ga & 0xFFFFFFFFu);         // global_addr[31:0]
  g0[3] = (unsigned int)((ga >> 32) & 0x01FFFFFFu)  // global_addr[56:32]
          | (2u << 30);                             // type = 2 ("image")
  i32x8 g1;
  g1[0] = (int)(1u << 16);                          // wg_mask=0, data_size=1 (2B)
  g1[1] = (int)((tensor_d0 & 0xFFFFu) << 16);       // tensor_dim0[15:0]
  g1[2] = (int)(((tensor_d0 >> 16) & 0xFFFFu) | ((tensor_d1 & 0xFFFFu) << 16));
  g1[3] = (int)(((tensor_d1 >> 16) & 0xFFFFu) | ((tile_d0 & 0xFFFFu) << 16));
  g1[4] = (int)(tile_d1 & 0xFFFFu);                 // tile_dim1 ; tile_dim2=0
  g1[5] = (int)(stride0 & 0xFFFFFFFFull);           // tensor_dim0_stride[31:0]
  g1[6] = (int)((stride0 >> 32) & 0xFFFFull);       // stride0[47:32]; dim1_stride=0
  g1[7] = 0;
  i32x4 z4 = (i32x4)0;
#if __clang_major__ >= 23
  i32x8 z8 = (i32x8)0;
  __builtin_amdgcn_tensor_load_to_lds(g0, g1, z4, z4, z8, 0);
#else
  __builtin_amdgcn_tensor_load_to_lds(g0, g1, z4, z4, 0);
#endif
}

// ---------------------------------------------------------------------------
// Projection y = x @ W^T  (x: [16384,256] f32, Wbf: [256,256] bf16).
// Row-major bf16 output variant. Block = 8 waves, wave = 16 rows x 256 cols.
// ---------------------------------------------------------------------------
__global__ __launch_bounds__(256)
void proj_rm_kernel(const float* __restrict__ xb,
                    const unsigned short* __restrict__ wbf,
                    unsigned short* __restrict__ yb)
{
  const int lane = threadIdx.x & 31;
  const int wv   = threadIdx.x >> 5;
  const int hl   = lane >> 4;
  const int ln   = lane & 15;
  const int rowbase = blockIdx.x * 128 + wv * 16;

  AF a[8];
  const int arow = rowbase + ln;
  #pragma unroll
  for (int kc = 0; kc < 8; ++kc)
    #pragma unroll
    for (int i = 0; i < 16; ++i) {
      int k = i + hl * 16;   // contiguous k layout
      a[kc].u[i] = f2bf(xb[arow * 256 + kc * 32 + k]);
    }

  #pragma unroll 1
  for (int nt = 0; nt < 16; ++nt) {
    CF c;
    #pragma unroll
    for (int j = 0; j < 8; ++j) c.f[j] = 0.0f;
    const int ncol = nt * 16 + ln;
    #pragma unroll
    for (int kc = 0; kc < 8; ++kc) {
      AF bf;
      const unsigned short* p = wbf + ncol * 256 + kc * 32 + hl * 16;
      #pragma unroll
      for (int u = 0; u < 8; ++u) bf.d[u] = *(const unsigned int*)(p + 2 * u);
      c.v = __builtin_amdgcn_wmma_f32_16x16x32_bf16(
              false, a[kc].v, false, bf.v, (short)0, c.v, false, false);
    }
    #pragma unroll
    for (int j = 0; j < 8; ++j)
      yb[(rowbase + hl * 8 + j) * 256 + ncol] = f2bf(c.f[j]);
  }
}

// Transposed-per-batch output variant (for V): ytb[b][d][row], [4][256][4096].
__global__ __launch_bounds__(256)
void proj_tr_kernel(const float* __restrict__ xb,
                    const unsigned short* __restrict__ wbf,
                    unsigned short* __restrict__ ytb)
{
  const int lane = threadIdx.x & 31;
  const int wv   = threadIdx.x >> 5;
  const int hl   = lane >> 4;
  const int ln   = lane & 15;
  const int rowbase = blockIdx.x * 128 + wv * 16;

  AF a[8];
  const int arow = rowbase + ln;
  #pragma unroll
  for (int kc = 0; kc < 8; ++kc)
    #pragma unroll
    for (int i = 0; i < 16; ++i) {
      int k = i + hl * 16;
      a[kc].u[i] = f2bf(xb[arow * 256 + kc * 32 + k]);
    }

  #pragma unroll 1
  for (int nt = 0; nt < 16; ++nt) {
    CF c;
    #pragma unroll
    for (int j = 0; j < 8; ++j) c.f[j] = 0.0f;
    const int ncol = nt * 16 + ln;
    #pragma unroll
    for (int kc = 0; kc < 8; ++kc) {
      AF bf;
      const unsigned short* p = wbf + ncol * 256 + kc * 32 + hl * 16;
      #pragma unroll
      for (int u = 0; u < 8; ++u) bf.d[u] = *(const unsigned int*)(p + 2 * u);
      c.v = __builtin_amdgcn_wmma_f32_16x16x32_bf16(
              false, a[kc].v, false, bf.v, (short)0, c.v, false, false);
    }
    #pragma unroll
    for (int j = 0; j < 8; ++j) {
      int grow = rowbase + hl * 8 + j;
      int b = grow >> 12, lrow = grow & 4095;
      ytb[((size_t)(b * 256 + ncol)) * 4096 + lrow] = f2bf(c.f[j]);
    }
  }
}

__global__ __launch_bounds__(256)
void cvt_bf16_kernel(const float* __restrict__ src,
                     unsigned short* __restrict__ dst, int n)
{
  int i = blockIdx.x * blockDim.x + threadIdx.x;
  if (i < n) dst[i] = f2bf(src[i]);
}

// ---------------------------------------------------------------------------
// Flash attention. One wave = 16 query rows; block = 8 waves sharing one batch.
// Per iteration (32 keys): TDM prefetches next K tile (32x256 bf16, contiguous)
// and next V^T tile (256x32 bf16, 2D) into the alternate LDS buffers while all
// waves run 16 score WMMAs + online softmax + 16 P@V WMMAs from current LDS.
// ---------------------------------------------------------------------------
__global__ __launch_bounds__(256)
void attn_kernel(const unsigned short* __restrict__ qb,   // [16384][256] bf16
                 const unsigned short* __restrict__ kb,   // [16384][256] bf16
                 const unsigned short* __restrict__ vtb,  // [4][256][4096] bf16
                 float* __restrict__ ctx,                 // [4][4096][256]
                 float* __restrict__ attn,                // [4][4096][4096]
                 float* __restrict__ stats)               // [16384][2]
{
  __shared__ unsigned short ktile[2][32 * 256];           // 2 x 16KB
  __shared__ unsigned short vtile[2][256 * 32];           // 2 x 16KB
  __shared__ unsigned short plds[8][16 * 32];             // 8KB P staging

  const int lane = threadIdx.x & 31;
  const int wv   = threadIdx.x >> 5;
  const int hl   = lane >> 4;
  const int ln   = lane & 15;
  const int g      = blockIdx.x * 8 + wv;                 // 0..1023
  const int b      = blockIdx.x >> 5;                     // 32 blocks per batch
  const int qblock = g & 255;
  const int qbase  = b * 4096 + qblock * 16;

  // Resident Q fragments (16 rows x 256)
  AF aq[8];
  const int arow = qbase + ln;
  #pragma unroll
  for (int kc = 0; kc < 8; ++kc) {
    const unsigned short* p = qb + (size_t)arow * 256 + kc * 32 + hl * 16;
    #pragma unroll
    for (int t = 0; t < 8; ++t) {
      int k = 2 * t;
      aq[kc].d[t] = *(const unsigned int*)(p + k);
    }
  }

  CF O[16];
  #pragma unroll
  for (int n = 0; n < 16; ++n)
    #pragma unroll
    for (int j = 0; j < 8; ++j) O[n].f[j] = 0.0f;
  float m[8], s[8];
  #pragma unroll
  for (int j = 0; j < 8; ++j) { m[j] = -1e30f; s[j] = 0.0f; }

  const float scale = 0.0625f;
  float* attnRow = attn + (size_t)b * 4096 * 4096 + (size_t)(qblock * 16) * 4096;

  // Prologue: wave 0 DMAs tile 0 into buffer 0.
  if (wv == 0) {
    tdm_load_tile_bf16((unsigned int)(uintptr_t)&ktile[0][0],
                       kb + (size_t)(b * 4096) * 256,
                       8192, 1, 8192, 1, 8192ull);        // 1D contiguous 16KB
    tdm_load_tile_bf16((unsigned int)(uintptr_t)&vtile[0][0],
                       vtb + (size_t)b * 256 * 4096,
                       32, 256, 4096, 256, 4096ull);      // 2D: 256 rows of 32
  }

  #pragma unroll 1
  for (int jt = 0; jt < 128; ++jt) {
    const int buf = jt & 1;
    __builtin_amdgcn_s_wait_tensorcnt(0);   // current buffers are filled
    __syncthreads();                        // publish TDM-written LDS to block

    if (wv == 0 && jt + 1 < 128) {          // prefetch next tiles -> alt buffer
      const int kb2 = (jt + 1) * 32;
      tdm_load_tile_bf16((unsigned int)(uintptr_t)&ktile[buf ^ 1][0],
                         kb + (size_t)(b * 4096 + kb2) * 256,
                         8192, 1, 8192, 1, 8192ull);
      tdm_load_tile_bf16((unsigned int)(uintptr_t)&vtile[buf ^ 1][0],
                         vtb + (size_t)b * 256 * 4096 + kb2,
                         32, 256, 4096, 256, 4096ull);
    }

    // ---- scores: two 16x16 tiles over D=256 ----
    CF sc[2];
    #pragma unroll
    for (int t = 0; t < 2; ++t) {
      CF c;
      #pragma unroll
      for (int j = 0; j < 8; ++j) c.f[j] = 0.0f;
      const unsigned short* kl = &ktile[buf][(t * 16 + ln) * 256 + hl * 16];
      #pragma unroll
      for (int kc = 0; kc < 8; ++kc) {
        AF bf;
        const unsigned short* p = kl + kc * 32;
        #pragma unroll
        for (int u = 0; u < 8; ++u) bf.d[u] = *(const unsigned int*)(p + 2 * u);
        c.v = __builtin_amdgcn_wmma_f32_16x16x32_bf16(
                false, aq[kc].v, false, bf.v, (short)0, c.v, false, false);
      }
      const int kcolbase = jt * 32 + t * 16;
      #pragma unroll
      for (int j = 0; j < 8; ++j) {
        c.f[j] *= scale;
        attnRow[(size_t)(hl * 8 + j) * 4096 + kcolbase + ln] = c.f[j];  // raw
      }
      sc[t] = c;
    }

    // ---- online softmax (per row, reduced over 16-lane half) ----
    float corr[8];
    #pragma unroll
    for (int j = 0; j < 8; ++j) {
      float tm = rowmax16(fmaxf(sc[0].f[j], sc[1].f[j]));
      float mn = fmaxf(m[j], tm);
      corr[j]  = __expf(m[j] - mn);
      float p0 = __expf(sc[0].f[j] - mn);
      float p1 = __expf(sc[1].f[j] - mn);
      s[j] = s[j] * corr[j] + rowsum16(p0 + p1);
      m[j] = mn;
      sc[0].f[j] = p0; sc[1].f[j] = p1;
    }
    #pragma unroll
    for (int n = 0; n < 16; ++n)
      #pragma unroll
      for (int j = 0; j < 8; ++j) O[n].f[j] *= corr[j];

    // ---- stage P (16x32) through LDS into WMMA A layout ----
    __syncthreads();
    #pragma unroll
    for (int t = 0; t < 2; ++t)
      #pragma unroll
      for (int j = 0; j < 8; ++j)
        plds[wv][(hl * 8 + j) * 32 + t * 16 + ln] = f2bf(sc[t].f[j]);
    __syncthreads();
    AF ap;
    {
      const unsigned short* p = &plds[wv][ln * 32 + hl * 16];
      #pragma unroll
      for (int t = 0; t < 8; ++t) {
        int k = 2 * t;
        ap.d[t] = *(const unsigned int*)(p + k);
      }
    }

    // ---- O += P @ V from LDS V tile ([d=256][k=32]) ----
    #pragma unroll 1
    for (int n = 0; n < 16; ++n) {
      AF bv;
      const unsigned short* p = &vtile[buf][(n * 16 + ln) * 32 + hl * 16];
      #pragma unroll
      for (int u = 0; u < 8; ++u) bv.d[u] = *(const unsigned int*)(p + 2 * u);
      O[n].v = __builtin_amdgcn_wmma_f32_16x16x32_bf16(
                 false, ap.v, false, bv.v, (short)0, O[n].v, false, false);
    }
  }

  // ---- finalize ----
  #pragma unroll
  for (int j = 0; j < 8; ++j) {
    float inv = 1.0f / s[j];
    int qr = qbase + hl * 8 + j;
    #pragma unroll
    for (int n = 0; n < 16; ++n)
      ctx[(size_t)qr * 256 + n * 16 + ln] = O[n].f[j] * inv;
    if (ln == 0) { stats[qr * 2 + 0] = m[j]; stats[qr * 2 + 1] = s[j]; }
  }
}

// In-place normalize of raw scores: attn = exp(s - max) / sum
__global__ __launch_bounds__(256)
void norm_kernel(float* __restrict__ attn, const float* __restrict__ stats)
{
  size_t i = (size_t)blockIdx.x * blockDim.x + threadIdx.x;
  const size_t total  = (size_t)16384 * 4096;
  const size_t stride = (size_t)gridDim.x * blockDim.x;
  for (; i < total; i += stride) {
    int row = (int)(i >> 12);
    float mmax = stats[row * 2 + 0];
    float ssum = stats[row * 2 + 1];
    attn[i] = __expf(attn[i] - mmax) / ssum;
  }
}

extern "C" void kernel_launch(void* const* d_in, const int* in_sizes, int n_in,
                              void* d_out, int out_size, void* d_ws, size_t ws_size,
                              hipStream_t stream) {
  (void)in_sizes; (void)n_in; (void)out_size; (void)ws_size;
  const float* Q  = (const float*)d_in[0];
  const float* K  = (const float*)d_in[1];
  const float* V  = (const float*)d_in[2];
  const float* Wq = (const float*)d_in[3];
  const float* Wk = (const float*)d_in[4];
  const float* Wv = (const float*)d_in[5];

  float* ctx  = (float*)d_out;                       // [4][4096][256]
  float* attn = ctx + (size_t)4 * 4096 * 256;        // [4][4096][4096]

  // Workspace: bf16 Q, K, V^T (8MB each), bf16 weights (3x128KB), f32 stats
  unsigned short* qb  = (unsigned short*)d_ws;
  unsigned short* kb  = qb  + (size_t)16384 * 256;
  unsigned short* vtb = kb  + (size_t)16384 * 256;
  unsigned short* wqb = vtb + (size_t)16384 * 256;
  unsigned short* wkb = wqb + (size_t)256 * 256;
  unsigned short* wvb = wkb + (size_t)256 * 256;
  float*          st  = (float*)(wvb + (size_t)256 * 256);

  // Pre-convert weights to bf16 (tiny, L2-resident afterwards)
  cvt_bf16_kernel<<<256, 256, 0, stream>>>(Wq, wqb, 65536);
  cvt_bf16_kernel<<<256, 256, 0, stream>>>(Wk, wkb, 65536);
  cvt_bf16_kernel<<<256, 256, 0, stream>>>(Wv, wvb, 65536);

  // Projections (fused f32->bf16 A fragments), 128 blocks each
  proj_rm_kernel<<<128, 256, 0, stream>>>(Q, wqb, qb);
  proj_rm_kernel<<<128, 256, 0, stream>>>(K, wkb, kb);
  proj_tr_kernel<<<128, 256, 0, stream>>>(V, wvb, vtb);

  // Attention: 1024 query blocks of 16 rows -> 128 blocks x 8 waves
  attn_kernel<<<128, 256, 0, stream>>>(qb, kb, vtb, ctx, attn, st);

  // Normalize materialized attention
  norm_kernel<<<16384, 256, 0, stream>>>(attn, st);
}
